// AttentionBlock_5643587027409
// MI455X (gfx1250) — matrix operand order repaired and checked by reading, not compiled, not run on hardware
//
#include <hip/hip_runtime.h>
#include <hip/hip_bf16.h>
#include <math.h>

typedef __attribute__((ext_vector_type(16))) __bf16 v16bf;
typedef __attribute__((ext_vector_type(8)))  float  v8f;
typedef __attribute__((ext_vector_type(4)))  int    v4i;

#define EMBED  1024
#define HEADS  16
#define DKDIM  64
#define SEQ    2048
#define NBATCH 2
#define MROWS  4096   /* NBATCH*SEQ */
#define DFF    4096

#define AS1 __attribute__((address_space(1)))
#define AS3 __attribute__((address_space(3)))

#if __has_builtin(__builtin_amdgcn_global_load_async_to_lds_b128) && \
    __has_builtin(__builtin_amdgcn_s_wait_asynccnt)
#define USE_ASYNC_LDS 1
#else
#define USE_ASYNC_LDS 0
#endif

// ---------------------------------------------------------------------------
// fp32 -> bf16 (round to nearest even)
// ---------------------------------------------------------------------------
__device__ __forceinline__ unsigned short f32_bf16(float f) {
    unsigned u = __float_as_uint(f);
    unsigned r = u + 0x7FFFu + ((u >> 16) & 1u);
    if ((u & 0x7F800000u) == 0x7F800000u) r = u;   // inf/nan: truncate
    return (unsigned short)(r >> 16);
}

union AB32 { v16bf v; uint4 q[2]; unsigned short u[16]; };

// ---------------------------------------------------------------------------
// Weight conversion: W[K][N] fp32 row-major -> B-fragment-swizzled bf16:
//   Bsw[((k>>4)*N + n)*16 + (k&15)]
// so a lane's B fragment (k = kb + half*16 + j, fixed n) is 16 contiguous bf16.
// ---------------------------------------------------------------------------
__global__ void __launch_bounds__(256)
cvt_weight_sw(const float* __restrict__ W, unsigned short* __restrict__ Bsw,
              int K, int N) {
    size_t tid = (size_t)blockIdx.x * blockDim.x + threadIdx.x;
    if (tid >= (size_t)K * N) return;
    int k = (int)(tid / N);
    int n = (int)(tid % N);
    Bsw[((size_t)(k >> 4) * N + n) * 16 + (k & 15)] = f32_bf16(W[tid]);
}

// ---------------------------------------------------------------------------
// LayerNorm over last dim (1024), one block per row, bf16 output.
// ---------------------------------------------------------------------------
__global__ void __launch_bounds__(256)
ln_bf16(const float* __restrict__ x, const float* __restrict__ g,
        const float* __restrict__ b, unsigned short* __restrict__ out) {
    __shared__ float s1[256];
    __shared__ float s2[256];
    int row = blockIdx.x;
    int t   = threadIdx.x;
    const float4 v = ((const float4*)(x + (size_t)row * EMBED))[t];
    s1[t] = v.x + v.y + v.z + v.w;
    s2[t] = v.x*v.x + v.y*v.y + v.z*v.z + v.w*v.w;
    __syncthreads();
    for (int o = 128; o > 0; o >>= 1) {
        if (t < o) { s1[t] += s1[t + o]; s2[t] += s2[t + o]; }
        __syncthreads();
    }
    float mu  = s1[0] * (1.0f / EMBED);
    float var = s2[0] * (1.0f / EMBED) - mu * mu;
    float rs  = rsqrtf(var + 1e-5f);
    const float4 gg = ((const float4*)g)[t];
    const float4 bb = ((const float4*)b)[t];
    unsigned long long p = 0;
    p |= (unsigned long long)f32_bf16((v.x - mu) * rs * gg.x + bb.x);
    p |= (unsigned long long)f32_bf16((v.y - mu) * rs * gg.y + bb.y) << 16;
    p |= (unsigned long long)f32_bf16((v.z - mu) * rs * gg.z + bb.z) << 32;
    p |= (unsigned long long)f32_bf16((v.w - mu) * rs * gg.w + bb.w) << 48;
    *(unsigned long long*)(out + (size_t)row * EMBED + t * 4) = p;
}

// ---------------------------------------------------------------------------
// WMMA GEMM with LDS-staged, double-buffered B tiles.
// Block = 8 waves covering 128 rows x 64 cols; all waves share one B strip.
// B stage: 32(K) x 64(N) bf16 = 4 KB per buffer, copied cooperatively with
// GLOBAL_LOAD_ASYNC_TO_LDS_B128 (ASYNCcnt) when available.
// Epilogue fuses bias / exact GELU / residual; stores fp32, bf16, bf16^T.
// ---------------------------------------------------------------------------
template<bool GELU, bool RESID, bool STF32, bool STBF16, bool STBF16T>
__global__ void __launch_bounds__(256)
gemm_wmma(const unsigned short* __restrict__ A,
          const unsigned short* __restrict__ Bsw,
          const float* __restrict__ bias,
          const float* __restrict__ resid,
          float* __restrict__ outF,
          unsigned short* __restrict__ outB,
          unsigned short* __restrict__ outBT,
          int M, int N, int K, int ldT) {
    // sbuf[buf][kk(16-row block)][n (64 cols)][16 k-elems]
    __shared__ __align__(16) unsigned short sbuf[2][2][64][16];

    int n64 = N >> 6;
    int bm  = blockIdx.x / n64;
    int cb  = (blockIdx.x % n64) << 6;
    int w   = threadIdx.x >> 5;
    int tm  = bm * 8 + w;                 // 16-row tile index
    int lane = threadIdx.x & 31;
    int l16  = lane & 15;
    int half = lane >> 4;

    // cooperative B stage: 256 threads x 16B = 4 KB
    int st  = threadIdx.x;
    int skk = st >> 7;          // which 16-row k-block
    int sn  = (st >> 1) & 63;   // column within strip
    int sp  = st & 1;           // 8-short part
    auto stage = [&](int buf, int kb) {
        const unsigned short* gp =
            Bsw + ((size_t)((kb >> 4) + skk) * N + cb + sn) * 16 + sp * 8;
        unsigned short* lp = &sbuf[buf][skk][sn][sp * 8];
#if USE_ASYNC_LDS
        __builtin_amdgcn_global_load_async_to_lds_b128((AS1 v4i*)gp,
                                                       (AS3 v4i*)lp, 0, 0);
#else
        *(uint4*)lp = *(const uint4*)gp;
#endif
    };

    v8f acc[4];
#pragma unroll
    for (int c = 0; c < 4; ++c)
#pragma unroll
        for (int r = 0; r < 8; ++r) acc[c][r] = 0.0f;

    const unsigned short* Ar = A + (size_t)(tm * 16 + l16) * K;

    stage(0, 0);
    int nk = K >> 5;
    for (int i = 0; i < nk; ++i) {
        int kb = i << 5;
#if USE_ASYNC_LDS
        __builtin_amdgcn_s_wait_asynccnt(0);
#endif
        __syncthreads();                       // stage i visible to all waves
        if (i + 1 < nk) stage((i + 1) & 1, kb + 32);  // prefetch next buffer

        AB32 af;   // A 16x32 bf16 fragment: two contiguous 8-elem chunks
        af.q[0] = *(const uint4*)(Ar + kb + half * 16);
        af.q[1] = *(const uint4*)(Ar + kb + half * 16 + 8);
        int buf = i & 1;
#pragma unroll
        for (int c = 0; c < 4; ++c) {
            const unsigned short* bp = &sbuf[buf][half][c * 16 + l16][0];
            AB32 bf;
            bf.q[0] = *(const uint4*)(bp);
            bf.q[1] = *(const uint4*)(bp + 8);
            acc[c] = __builtin_amdgcn_wmma_f32_16x16x32_bf16(
                false, af.v, false, bf.v, (short)0, acc[c], false, false);
        }
        __syncthreads();                       // done reading buf before reuse
    }

#pragma unroll
    for (int c = 0; c < 4; ++c) {
        int col = cb + c * 16 + l16;
        float bi = bias[col];
#pragma unroll
        for (int r = 0; r < 8; ++r) {
            int row = tm * 16 + half * 8 + r;
            float v = acc[c][r] + bi;
            if constexpr (GELU)
                v = 0.5f * v * (1.0f + erff(v * 0.70710678118654752f));
            if constexpr (RESID) v += resid[(size_t)row * N + col];
            if constexpr (STF32)  outF[(size_t)row * N + col] = v;
            if constexpr (STBF16) outB[(size_t)row * N + col] = f32_bf16(v);
            if constexpr (STBF16T) outBT[(size_t)col * ldT + row] = f32_bf16(v);
        }
    }
    (void)M;
}

// ---------------------------------------------------------------------------
// Causal flash attention, one wave per (batch, head, 16-query tile).
// Q,K bf16 row-major [4096][1024]; Vt bf16 transposed [1024][4096].
// Output bf16 row-major [4096][1024].
// ---------------------------------------------------------------------------
__global__ void __launch_bounds__(256)
attn_wmma(const unsigned short* __restrict__ Q,
          const unsigned short* __restrict__ Km,
          const unsigned short* __restrict__ Vt,
          unsigned short* __restrict__ O) {
    __shared__ __align__(16) unsigned short pbuf[8][16 * 32];
    int wv   = threadIdx.x >> 5;
    int wid  = blockIdx.x * (blockDim.x >> 5) + wv;
    int qt   = wid & 127;
    int h    = (wid >> 7) & 15;
    int b    = wid >> 11;
    int lane = threadIdx.x & 31;
    int l16  = lane & 15;
    int half = lane >> 4;
    int q0   = qt * 16;

    // Q fragments (A-layout), fixed for the whole key loop
    const unsigned short* qp =
        Q + (size_t)(b * SEQ + q0 + l16) * EMBED + h * DKDIM;
    AB32 qf0, qf1;
    qf0.q[0] = *(const uint4*)(qp + 0  + half * 16);
    qf0.q[1] = *(const uint4*)(qp + 8  + half * 16);
    qf1.q[0] = *(const uint4*)(qp + 32 + half * 16);
    qf1.q[1] = *(const uint4*)(qp + 40 + half * 16);

    float m[8], l[8];
    v8f o[4];
#pragma unroll
    for (int r = 0; r < 8; ++r) { m[r] = -3e38f; l[r] = 0.0f; }
#pragma unroll
    for (int c = 0; c < 4; ++c)
#pragma unroll
        for (int r = 0; r < 8; ++r) o[c][r] = 0.0f;

    unsigned short* P = pbuf[wv];

    for (int kb = 0; kb <= q0 + 15; kb += 32) {
        // --- scores: S[16q x 32k] as two 16-col WMMA tiles, contraction dk=64
        v8f s[2];
#pragma unroll
        for (int t = 0; t < 2; ++t) {
#pragma unroll
            for (int r = 0; r < 8; ++r) s[t][r] = 0.0f;
            const unsigned short* kp =
                Km + (size_t)(b * SEQ + kb + t * 16 + l16) * EMBED + h * DKDIM
                   + half * 16;
            AB32 b0, b1;
            b0.q[0] = *(const uint4*)(kp);
            b0.q[1] = *(const uint4*)(kp + 8);
            b1.q[0] = *(const uint4*)(kp + 32);
            b1.q[1] = *(const uint4*)(kp + 40);
            s[t] = __builtin_amdgcn_wmma_f32_16x16x32_bf16(
                false, qf0.v, false, b0.v, (short)0, s[t], false, false);
            s[t] = __builtin_amdgcn_wmma_f32_16x16x32_bf16(
                false, qf1.v, false, b1.v, (short)0, s[t], false, false);
        }
        // --- scale + causal mask (key > query -> -inf)
#pragma unroll
        for (int t = 0; t < 2; ++t) {
            int key = kb + t * 16 + l16;
#pragma unroll
            for (int r = 0; r < 8; ++r) {
                float v = s[t][r] * 0.125f;           // 1/sqrt(64)
                if (key > q0 + half * 8 + r) v = -3e38f;
                s[t][r] = v;
            }
        }
        // --- online softmax stats (rows live across the 16 lanes of a half)
        float pl0[8], pl1[8];
#pragma unroll
        for (int r = 0; r < 8; ++r) {
            float mx = fmaxf(s[0][r], s[1][r]);
            for (int off = 1; off < 16; off <<= 1)
                mx = fmaxf(mx, __shfl_xor(mx, off, 32));
            float mn = fmaxf(m[r], mx);
            float al = __expf(m[r] - mn);
            float p0 = __expf(s[0][r] - mn);
            float p1 = __expf(s[1][r] - mn);
            float sum = p0 + p1;
            for (int off = 1; off < 16; off <<= 1)
                sum += __shfl_xor(sum, off, 32);
            l[r] = l[r] * al + sum;
            m[r] = mn;
            pl0[r] = p0; pl1[r] = p1;
#pragma unroll
            for (int c = 0; c < 4; ++c) o[c][r] *= al;
        }
        // --- transpose P from C-layout to A-layout via per-wave LDS
#pragma unroll
        for (int r = 0; r < 8; ++r) {
            P[(half * 8 + r) * 32 + l16]      = f32_bf16(pl0[r]);
            P[(half * 8 + r) * 32 + 16 + l16] = f32_bf16(pl1[r]);
        }
        __asm__ volatile("" ::: "memory");   // DS ops are in-order per wave
        AB32 pf;
        pf.q[0] = *(const uint4*)(P + l16 * 32 + half * 16);
        pf.q[1] = *(const uint4*)(P + l16 * 32 + half * 16 + 8);
        __asm__ volatile("" ::: "memory");
        // --- O += P @ V  (V read from transposed layout: contiguous in key)
#pragma unroll
        for (int c = 0; c < 4; ++c) {
            const unsigned short* vq =
                Vt + (size_t)(h * DKDIM + c * 16 + l16) * MROWS
                   + b * SEQ + kb + half * 16;
            AB32 vf;
            vf.q[0] = *(const uint4*)(vq);
            vf.q[1] = *(const uint4*)(vq + 8);
            o[c] = __builtin_amdgcn_wmma_f32_16x16x32_bf16(
                false, pf.v, false, vf.v, (short)0, o[c], false, false);
        }
    }
    // --- normalize and store bf16 row-major
#pragma unroll
    for (int c = 0; c < 4; ++c) {
        int col = h * DKDIM + c * 16 + l16;
#pragma unroll
        for (int r = 0; r < 8; ++r) {
            int row = b * SEQ + q0 + half * 8 + r;
            O[(size_t)row * EMBED + col] = f32_bf16(o[c][r] / l[r]);
        }
    }
}

// ---------------------------------------------------------------------------
// Host-side orchestration
// ---------------------------------------------------------------------------
extern "C" void kernel_launch(void* const* d_in, const int* in_sizes, int n_in,
                              void* d_out, int out_size, void* d_ws,
                              size_t ws_size, hipStream_t stream) {
    (void)in_sizes; (void)n_in; (void)out_size; (void)ws_size;
    const float* x     = (const float*)d_in[0];
    const float* ln1_g = (const float*)d_in[1];
    const float* ln1_b = (const float*)d_in[2];
    const float* wq    = (const float*)d_in[3];
    const float* bq    = (const float*)d_in[4];
    const float* wk    = (const float*)d_in[5];
    const float* bk    = (const float*)d_in[6];
    const float* wv    = (const float*)d_in[7];
    const float* bv    = (const float*)d_in[8];
    const float* wo    = (const float*)d_in[9];
    const float* bo    = (const float*)d_in[10];
    const float* ln2_g = (const float*)d_in[11];
    const float* ln2_b = (const float*)d_in[12];
    const float* w1    = (const float*)d_in[13];
    const float* b1    = (const float*)d_in[14];
    const float* w2    = (const float*)d_in[15];
    const float* b2    = (const float*)d_in[16];
    float* out = (float*)d_out;

    char* ws = (char*)d_ws;
    size_t off = 0;
    auto alloc = [&](size_t bytes) {
        void* p = ws + off;
        off += (bytes + 255) & ~(size_t)255;
        return p;
    };
    const size_t W_DD  = (size_t)EMBED * EMBED * 2;   // 2 MB
    const size_t W_DF  = (size_t)EMBED * DFF * 2;     // 8 MB
    const size_t ACT_B = (size_t)MROWS * EMBED * 2;   // 8 MB
    unsigned short* wq_sw = (unsigned short*)alloc(W_DD);
    unsigned short* wk_sw = (unsigned short*)alloc(W_DD);
    unsigned short* wv_sw = (unsigned short*)alloc(W_DD);
    unsigned short* wo_sw = (unsigned short*)alloc(W_DD);
    unsigned short* w1_sw = (unsigned short*)alloc(W_DF);
    unsigned short* w2_sw = (unsigned short*)alloc(W_DF);
    unsigned short* xn1   = (unsigned short*)alloc(ACT_B);
    unsigned short* Qb    = (unsigned short*)alloc(ACT_B);
    unsigned short* Kb    = (unsigned short*)alloc(ACT_B);
    unsigned short* Vtb   = (unsigned short*)alloc(ACT_B);
    unsigned short* attnb = (unsigned short*)alloc(ACT_B);
    float*          x1    = (float*)alloc((size_t)MROWS * EMBED * 4);
    unsigned short* h2    = (unsigned short*)alloc(ACT_B);
    // FFN hidden (32 MB) aliases dead xn1/Q/K/Vt region (4 x 8 MB)
    unsigned short* ffh   = xn1;

    // 1) weights -> swizzled bf16
    cvt_weight_sw<<<(EMBED * EMBED + 255) / 256, 256, 0, stream>>>(wq, wq_sw, EMBED, EMBED);
    cvt_weight_sw<<<(EMBED * EMBED + 255) / 256, 256, 0, stream>>>(wk, wk_sw, EMBED, EMBED);
    cvt_weight_sw<<<(EMBED * EMBED + 255) / 256, 256, 0, stream>>>(wv, wv_sw, EMBED, EMBED);
    cvt_weight_sw<<<(EMBED * EMBED + 255) / 256, 256, 0, stream>>>(wo, wo_sw, EMBED, EMBED);
    cvt_weight_sw<<<(EMBED * DFF + 255) / 256, 256, 0, stream>>>(w1, w1_sw, EMBED, DFF);
    cvt_weight_sw<<<(EMBED * DFF + 255) / 256, 256, 0, stream>>>(w2, w2_sw, DFF, EMBED);

    // 2) LN1 -> bf16
    ln_bf16<<<MROWS, 256, 0, stream>>>(x, ln1_g, ln1_b, xn1);

    // 3) Q/K/V projections (V stored transposed for attention B-operand)
    //    grid = (M/128) * (N/64)
    gemm_wmma<false, false, false, true, false><<<512, 256, 0, stream>>>(
        xn1, wq_sw, bq, nullptr, nullptr, Qb, nullptr, MROWS, EMBED, EMBED, 0);
    gemm_wmma<false, false, false, true, false><<<512, 256, 0, stream>>>(
        xn1, wk_sw, bk, nullptr, nullptr, Kb, nullptr, MROWS, EMBED, EMBED, 0);
    gemm_wmma<false, false, false, false, true><<<512, 256, 0, stream>>>(
        xn1, wv_sw, bv, nullptr, nullptr, nullptr, Vtb, MROWS, EMBED, EMBED, MROWS);

    // 4) causal flash attention (2*16*128 = 4096 waves)
    attn_wmma<<<512, 256, 0, stream>>>(Qb, Kb, Vtb, attnb);

    // 5) output projection + residual -> x1 (fp32)
    gemm_wmma<false, true, true, false, false><<<512, 256, 0, stream>>>(
        attnb, wo_sw, bo, x, x1, nullptr, nullptr, MROWS, EMBED, EMBED, 0);

    // 6) LN2 -> bf16
    ln_bf16<<<MROWS, 256, 0, stream>>>(x1, ln2_g, ln2_b, h2);

    // 7) FFN GEMM1 + exact GELU -> bf16 hidden
    gemm_wmma<true, false, false, true, false><<<2048, 256, 0, stream>>>(
        h2, w1_sw, b1, nullptr, nullptr, ffh, nullptr, MROWS, DFF, EMBED, 0);

    // 8) FFN GEMM2 + residual -> final fp32 output
    gemm_wmma<false, true, true, false, false><<<512, 256, 0, stream>>>(
        ffh, w2_sw, b2, x1, out, nullptr, nullptr, MROWS, EMBED, DFF, 0);
}
